// AGLTSA_11751030522770
// MI455X (gfx1250) — matrix operand order repaired and checked by reading, not compiled, not run on hardware
//
#include <hip/hip_runtime.h>
#include <hip/hip_bf16.h>

// ---------------------------------------------------------------------------
// AGL + temporal-self-attention network for MI455X (gfx1250, wave32).
// All heavy GEMMs: f16 operands (pre-converted once into scratch, K-strides
// padded to multiples of 8 halves), f32 WMMA accumulate, async-to-LDS
// double-buffered tile pipeline:
//   global_load_async_to_lds_b128  ->  s_wait_asynccnt  ->  barrier  ->
//   issue next tile  ->  8x v_wmma_f32_16x16x32_f16  ->  barrier
// ---------------------------------------------------------------------------

typedef __attribute__((ext_vector_type(16))) _Float16 v16h;
typedef __attribute__((ext_vector_type(8)))  float    v8f;

#define BM 64
#define BN 128
#define BKT 32
#define GEMM_THREADS 128

// ---- CDNA5 async copy (ASYNCcnt) helpers: inline asm per CDNA5 ISA ----
__device__ __forceinline__ void async_copy_b128(void* lds_dst, const void* gsrc)
{
    unsigned lds_off = (unsigned)(unsigned long long)lds_dst;  // LDS aperture low bits
    asm volatile("global_load_async_to_lds_b128 %0, %1, off"
                 :: "v"(lds_off), "v"(gsrc)
                 : "memory");
}
__device__ __forceinline__ void async_wait_all()
{
    asm volatile("s_wait_asynccnt 0x0" ::: "memory");
}

// Generic batched GEMM: C = act(alpha * A@B + bias + Add)
//  A: (M x K) f16, row stride lda (halves, multiple of 8; rows zero-padded to KloadA)
//  B: (K x N) f16 row stride ldb; or transB: Bsrc is (N x K) row-major
//  Outputs: C32 (f32) and/or C16 (f16), same ldc; either may be null.
//  Batch: z -> (zo=z/inner, zi=z%inner) with per-operand strides.
__global__ __launch_bounds__(GEMM_THREADS)
void agltsa_wmma_gemm(const _Float16* __restrict__ Abase, const _Float16* __restrict__ Bbase,
                      float* __restrict__ C32base, _Float16* __restrict__ C16base,
                      const float* __restrict__ biasPtr, const float* __restrict__ Addbase,
                      int M, int N, int K, int KloadA, int NloadB,
                      long long lda, long long ldb, long long ldc, long long ldadd,
                      long long sAo, long long sAi, long long sBo, long long sBi,
                      long long sCo, long long sCi, long long sAddo, long long sAddi,
                      int innerBatch, float alpha, int relu, int transB,
                      const _Float16* __restrict__ zerobuf)
{
    __shared__ __align__(16) _Float16 As[2][BM][BKT + 8];   // 2 x 5KB
    __shared__ __align__(16) _Float16 Bs[2][BKT][BN + 8];   // 2 x 8.5KB

    const int z  = blockIdx.z;
    const int zo = z / innerBatch;
    const int zi = z - zo * innerBatch;
    const _Float16* A  = Abase + zo * sAo + zi * sAi;
    const _Float16* Bm = Bbase + zo * sBo + zi * sBi;
    float*    C32 = C32base ? (C32base + zo * sCo + zi * sCi) : (float*)0;
    _Float16* C16 = C16base ? (C16base + zo * sCo + zi * sCi) : (_Float16*)0;
    const float* Add = Addbase ? (Addbase + zo * sAddo + zi * sAddi) : (const float*)0;

    const int tid  = threadIdx.x;
    const int lane = tid & 31;
    const int wave = tid >> 5;
    const int wm   = wave >> 1;            // 0..1 : 32-row slab
    const int wn   = wave & 1;             // 0..1 : 64-col slab
    const int rowBlock = blockIdx.y * BM;
    const int colBlock = blockIdx.x * BN;

    v8f acc[2][4] = {};

    // stage tile kt into buffer sbuf (async for A and non-trans B)
    auto stage = [&](int k0, int sbuf) {
        // A tile: BM x BKT halves, 16B chunks
        #pragma unroll
        for (int i = tid * 8; i < BM * BKT; i += GEMM_THREADS * 8) {
            int r  = i >> 5;            // / BKT
            int kk = i & (BKT - 1);
            int gr = rowBlock + r;
            int gk = k0 + kk;
            const _Float16* src = (gr < M && gk < KloadA)
                                ? (A + (long long)gr * lda + gk) : zerobuf;
            async_copy_b128(&As[sbuf][r][kk], src);
        }
        if (!transB) {
            #pragma unroll
            for (int i = tid * 8; i < BKT * BN; i += GEMM_THREADS * 8) {
                int kk = i >> 7;        // / BN
                int nn = i & (BN - 1);
                int gk = k0 + kk;
                int gn = colBlock + nn;
                const _Float16* src = (gk < K && gn < NloadB)
                                    ? (Bm + (long long)gk * ldb + gn) : zerobuf;
                async_copy_b128(&Bs[sbuf][kk][nn], src);
            }
        } else {
            // logical B[k][n] = Bsrc[n][k]; synchronous scatter into [k][n] layout
            #pragma unroll
            for (int i = tid * 8; i < BKT * BN; i += GEMM_THREADS * 8) {
                int nn = i >> 5;        // / BKT
                int kk = i & (BKT - 1);
                int gn = colBlock + nn;
                int gk = k0 + kk;
                float4 raw = make_float4(0.f, 0.f, 0.f, 0.f);
                if (gn < N && gk < K)
                    raw = *(const float4*)(Bm + (long long)gn * ldb + gk);  // 8 halves
                const _Float16* h = (const _Float16*)&raw;
                #pragma unroll
                for (int t = 0; t < 8; ++t) Bs[sbuf][kk + t][nn] = h[t];
            }
        }
    };

    const int nk = (K + BKT - 1) / BKT;
    stage(0, 0);

    for (int kt = 0; kt < nk; ++kt) {
        const int buf = kt & 1;
        async_wait_all();          // own wave's async copies into `buf` done
        __syncthreads();           // all waves' copies visible
        if (kt + 1 < nk) stage((kt + 1) * BKT, buf ^ 1);   // overlap next copy

        // fragments per ISA VGPR layouts
        const int m16 = lane & 15;
        const int kb  = (lane & 16) ? 8 : 0;    // A 16-bit 16x32 K-split
        v16h af[2], bf[4];
        #pragma unroll
        for (int mi = 0; mi < 2; ++mi) {
            const _Float16* ap = &As[buf][wm * 32 + mi * 16 + m16][0];
            #pragma unroll
            for (int j = 0; j < 8; ++j) {
                af[mi][j]     = ap[kb + j];
                af[mi][8 + j] = ap[kb + 16 + j];
            }
        }
        #pragma unroll
        for (int ni = 0; ni < 4; ++ni) {
            const _Float16* bp = &Bs[buf][0][wn * 64 + ni * 16 + m16];   // lane = N col
            #pragma unroll
            for (int j = 0; j < 8; ++j) {
                bf[ni][j]     = bp[(kb + j) * (BN + 8)];
                bf[ni][8 + j] = bp[(kb + 16 + j) * (BN + 8)];
            }
        }
        #pragma unroll
        for (int mi = 0; mi < 2; ++mi)
            #pragma unroll
            for (int ni = 0; ni < 4; ++ni)
                acc[mi][ni] = __builtin_amdgcn_wmma_f32_16x16x32_f16(
                    false, af[mi], false, bf[ni], (short)0, acc[mi][ni], false, false);

        __syncthreads();           // everyone done reading `buf` before overwrite
    }

    // epilogue: C/D layout -> vgpr r is M=r (lane<16) else M=r+8, N=lane&15
    #pragma unroll
    for (int mi = 0; mi < 2; ++mi) {
        #pragma unroll
        for (int ni = 0; ni < 4; ++ni) {
            #pragma unroll
            for (int r = 0; r < 8; ++r) {
                int rowl = (lane < 16) ? r : r + 8;
                int coll = lane & 15;
                int gr = rowBlock + wm * 32 + mi * 16 + rowl;
                int gc = colBlock + wn * 64 + ni * 16 + coll;
                if (gr < M && gc < N) {
                    float v = acc[mi][ni][r] * alpha;
                    if (biasPtr) v += biasPtr[gc];
                    if (Add)     v += Add[(long long)gr * ldadd + gc];
                    if (relu)    v = fmaxf(v, 0.f);
                    if (C32) C32[(long long)gr * ldc + gc] = v;
                    if (C16) C16[(long long)gr * ldc + gc] = (_Float16)v;
                }
            }
        }
    }
}

// ---------------------------------------------------------------------------
// Small helper kernels
// ---------------------------------------------------------------------------

__global__ void agltsa_zero16(_Float16* __restrict__ p, int n)
{
    int i = blockIdx.x * blockDim.x + threadIdx.x;
    if (i < n) p[i] = (_Float16)0.f;
}

__global__ void agltsa_cvt(const float* __restrict__ src, _Float16* __restrict__ dst, long long n)
{
    long long i = (long long)blockIdx.x * blockDim.x + threadIdx.x;
    if (i < n) dst[i] = (_Float16)src[i];
}

// f32 (rows x cols) -> f16 (rows x ldpad), zero-filled pad columns
__global__ void agltsa_cvt_pad(const float* __restrict__ src, _Float16* __restrict__ dst,
                               int rows, int cols, int ldpad)
{
    long long i = (long long)blockIdx.x * blockDim.x + threadIdx.x;
    long long total = (long long)rows * ldpad;
    if (i >= total) return;
    int r = (int)(i / ldpad), c = (int)(i - (long long)r * ldpad);
    dst[i] = (c < cols) ? (_Float16)src[(long long)r * cols + c] : (_Float16)0.f;
}

// gram = emb @ emb^T, written as padded f16 (n x ldpad)
__global__ void agltsa_gram(const float* __restrict__ emb, _Float16* __restrict__ gram16,
                            int n, int e, int ldpad)
{
    long long i = (long long)blockIdx.x * blockDim.x + threadIdx.x;
    long long total = (long long)n * ldpad;
    if (i >= total) return;
    int r = (int)(i / ldpad), c = (int)(i - (long long)r * ldpad);
    float s = 0.f;
    if (c < n)
        for (int k = 0; k < e; ++k) s += emb[r * e + k] * emb[c * e + k];
    gram16[i] = (_Float16)s;
}

// row softmax in place (f32) + f16 copy with zero-padded columns
__global__ void agltsa_softmax_rows(float* __restrict__ X, _Float16* __restrict__ out16,
                                    int rows, int cols, int ld16)
{
    int row = blockIdx.x;
    if (row >= rows) return;
    float* p = X + (long long)row * cols;
    __shared__ float red[256];
    float mx = -3.402823466e+38f;
    for (int c = threadIdx.x; c < cols; c += blockDim.x) mx = fmaxf(mx, p[c]);
    red[threadIdx.x] = mx; __syncthreads();
    for (int s = blockDim.x >> 1; s > 0; s >>= 1) {
        if (threadIdx.x < s) red[threadIdx.x] = fmaxf(red[threadIdx.x], red[threadIdx.x + s]);
        __syncthreads();
    }
    float m = red[0]; __syncthreads();
    float sum = 0.f;
    for (int c = threadIdx.x; c < cols; c += blockDim.x) {
        float e = __expf(p[c] - m);
        p[c] = e; sum += e;
    }
    red[threadIdx.x] = sum; __syncthreads();
    for (int s = blockDim.x >> 1; s > 0; s >>= 1) {
        if (threadIdx.x < s) red[threadIdx.x] += red[threadIdx.x + s];
        __syncthreads();
    }
    float inv = 1.f / red[0];
    for (int c = threadIdx.x; c < cols; c += blockDim.x) p[c] *= inv;
    if (out16) {
        _Float16* q = out16 + (long long)row * ld16;
        for (int c = threadIdx.x; c < ld16; c += blockDim.x)
            q[c] = (c < cols) ? (_Float16)p[c] : (_Float16)0.f;
    }
}

// T2f32 holds 2*(A@A); emit f16 padded copy of 2*A@A - I
__global__ void agltsa_cheb_fix(const float* __restrict__ T2, _Float16* __restrict__ out16,
                                int n, int ldpad)
{
    long long i = (long long)blockIdx.x * blockDim.x + threadIdx.x;
    long long total = (long long)n * ldpad;
    if (i >= total) return;
    int r = (int)(i / ldpad), c = (int)(i - (long long)r * ldpad);
    float v = 0.f;
    if (c < n) v = T2[(long long)r * n + c] - ((r == c) ? 1.f : 0.f);
    out16[i] = (_Float16)v;
}

// x (BT,N,C) f32 -> Xt16 (N, BT*C) f16
__global__ void agltsa_transpose_x(const float* __restrict__ x, _Float16* __restrict__ Xt,
                                   int BT, int N, int C)
{
    long long idx = (long long)blockIdx.x * blockDim.x + threadIdx.x;
    long long total = (long long)BT * N * C;
    if (idx >= total) return;
    int bt = (int)(idx / (N * C));
    int r  = (int)(idx - (long long)bt * (N * C));
    int m = r / C, c = r - m * C;
    Xt[(long long)m * ((long long)BT * C) + (long long)bt * C + c] = (_Float16)x[idx];
}

// g[bt][n*4+d] = sum_{k,c} Yk[n][bt*4+c]*W[(k*4+c)*4+d] + bias[d] (+pos enc)
// Y0 term (S0 = I) read straight from x.
__global__ void agltsa_glout(const float* __restrict__ xsrc,
                             const float* __restrict__ Y1, const float* __restrict__ Y2,
                             const float* __restrict__ W, const float* __restrict__ bias,
                             float* __restrict__ g, int BT, int N, int T, int addpos)
{
    long long idx = (long long)blockIdx.x * blockDim.x + threadIdx.x;
    long long total = (long long)BT * N;
    if (idx >= total) return;
    int bt = (int)(idx / N);
    int n  = (int)(idx - (long long)bt * N);
    long long ld  = (long long)BT * 4;
    long long col = (long long)bt * 4;
    float xv[12];
    #pragma unroll
    for (int c = 0; c < 4; ++c) {
        xv[c]     = xsrc[(long long)bt * (N * 4) + (long long)n * 4 + c];
        xv[4 + c] = Y1[(long long)n * ld + col + c];
        xv[8 + c] = Y2[(long long)n * ld + col + c];
    }
    float* out = g + (long long)bt * (N * 4) + (long long)n * 4;
    int t = bt % T;
    const float inc = 9.210340372f / 599.0f;
    #pragma unroll
    for (int d = 0; d < 4; ++d) {
        float s = bias[d];
        #pragma unroll
        for (int j = 0; j < 12; ++j) s += xv[j] * W[j * 4 + d];
        if (addpos) {
            int hh = n * 4 + d;
            s += (hh < 600) ? __sinf((float)t * __expf(-inc * (float)hh))
                            : __cosf((float)t * __expf(-inc * (float)(hh - 600)));
        }
        out[d] = s;
    }
}

__global__ void agltsa_layernorm(const float* __restrict__ X, const float* __restrict__ gw,
                                 const float* __restrict__ bw,
                                 float* __restrict__ Y32, _Float16* __restrict__ Y16, int cols)
{
    int row = blockIdx.x;
    const float* x = X + (long long)row * cols;
    __shared__ float s1[256], s2[256];
    float a = 0.f, b = 0.f;
    for (int c = threadIdx.x; c < cols; c += blockDim.x) {
        float v = x[c]; a += v; b += v * v;
    }
    s1[threadIdx.x] = a; s2[threadIdx.x] = b; __syncthreads();
    for (int s = blockDim.x >> 1; s > 0; s >>= 1) {
        if (threadIdx.x < s) { s1[threadIdx.x] += s1[threadIdx.x + s]; s2[threadIdx.x] += s2[threadIdx.x + s]; }
        __syncthreads();
    }
    float mean = s1[0] / (float)cols;
    float var  = s2[0] / (float)cols - mean * mean;
    float inv  = rsqrtf(var + 1e-6f);
    for (int c = threadIdx.x; c < cols; c += blockDim.x) {
        float r = (x[c] - mean) * inv * gw[c] + bw[c];
        if (Y32) Y32[(long long)row * cols + c] = r;
        if (Y16) Y16[(long long)row * cols + c] = (_Float16)r;
    }
}

// ---------------------------------------------------------------------------
// Host orchestration
// ---------------------------------------------------------------------------

static inline void launch_gemm(hipStream_t stream,
    const _Float16* A, const _Float16* Bm, float* C32, _Float16* C16,
    const float* bias, const float* Add,
    int M, int N, int K, int KloadA, int NloadB,
    long long lda, long long ldb, long long ldc, long long ldadd,
    long long sAo, long long sAi, long long sBo, long long sBi,
    long long sCo, long long sCi, long long sAddo, long long sAddi,
    int nbatch, int inner, float alpha, int relu, int transB,
    const _Float16* zerobuf)
{
    dim3 grid((N + BN - 1) / BN, (M + BM - 1) / BM, nbatch);
    agltsa_wmma_gemm<<<grid, GEMM_THREADS, 0, stream>>>(
        A, Bm, C32, C16, bias, Add, M, N, K, KloadA, NloadB,
        lda, ldb, ldc, ldadd, sAo, sAi, sBo, sBi, sCo, sCi, sAddo, sAddi,
        inner, alpha, relu, transB, zerobuf);
}

static inline void launch_gemm2d(hipStream_t stream,
    const _Float16* A, const _Float16* Bm, float* C32, _Float16* C16,
    const float* bias, const float* Add,
    int M, int N, int K, int KloadA, int NloadB,
    long long lda, long long ldb, long long ldc, long long ldadd,
    float alpha, int relu, const _Float16* zerobuf)
{
    launch_gemm(stream, A, Bm, C32, C16, bias, Add, M, N, K, KloadA, NloadB,
                lda, ldb, ldc, ldadd, 0,0,0,0,0,0,0,0, 1, 1, alpha, relu, 0, zerobuf);
}

extern "C" void kernel_launch(void* const* d_in, const int* in_sizes, int n_in,
                              void* d_out, int out_size, void* d_ws, size_t ws_size,
                              hipStream_t stream)
{
    (void)in_sizes; (void)n_in; (void)out_size; (void)ws_size;
    const int Bb = 32, T = 128, Nn = 300, Cc = 4, L = 2, Hh = 1200, Ff = 4800;
    const int HEADS = 6, DK = 200, E = 10;
    const int BT = Bb * T;              // 4096
    const int KC = BT * Cc;             // 16384
    const int NP = 304;                 // 300 padded to multiple of 8
    const float scale = 0.07071067812f; // dk^-0.5

    const float* inputs = (const float*)d_in[0];
    const float* emb    = (const float*)d_in[1];
    const float* aglW   = (const float*)d_in[2];
    const float* aglB   = (const float*)d_in[3];
    const float* aoW    = (const float*)d_in[4];
    const float* aoB    = (const float*)d_in[5];
    const float* Wq     = (const float*)d_in[6];
    const float* Wk     = (const float*)d_in[7];
    const float* Wv     = (const float*)d_in[8];
    const float* Wo     = (const float*)d_in[9];
    const float* ln1g   = (const float*)d_in[10];
    const float* ln1b   = (const float*)d_in[11];
    const float* W1     = (const float*)d_in[12];
    const float* b1     = (const float*)d_in[13];
    const float* W2     = (const float*)d_in[14];
    const float* b2     = (const float*)d_in[15];
    const float* ln2g   = (const float*)d_in[16];
    const float* ln2b   = (const float*)d_in[17];
    const float* lastg  = (const float*)d_in[18];
    const float* lastb  = (const float*)d_in[19];

    // ---- workspace carve-out (~300 MB assumed available) ----
    float* w = (float*)d_ws;
    size_t off = 0;                                    // in floats, 16-float granules
    auto allocf = [&](size_t n) { n = (n + 15) & ~(size_t)15; float* p = w + off; off += n; return p; };
    auto alloch = [&](size_t nh) {
        size_t nf = ((nh + 1) / 2 + 15) & ~(size_t)15;
        _Float16* p = (_Float16*)(w + off); off += nf; return p;
    };
    // f32 buffers
    float* Aadj = allocf((size_t)Nn * Nn);
    float* T2f  = allocf((size_t)Nn * Nn);
    float* Y1   = allocf((size_t)Nn * KC);
    float* Y2   = allocf((size_t)Nn * KC);
    float* gbuf = allocf((size_t)BT * Hh);
    float* g2   = allocf((size_t)BT * Hh);
    float* attf = allocf((size_t)Bb * HEADS * T * T);
    float* xbuf = allocf((size_t)BT * Hh);
    // f16 buffers (+64 slack for overshooting b128 chunks)
    _Float16* zb16    = alloch(64);
    _Float16* gram16  = alloch((size_t)Nn * NP + 64);
    _Float16* aglW16  = alloch((size_t)L * Nn * NP + 64);
    _Float16* Aadj16  = alloch((size_t)Nn * NP + 64);
    _Float16* T216    = alloch((size_t)Nn * NP + 64);
    _Float16* Xt16    = alloch((size_t)Nn * KC + 64);
    _Float16* y16     = alloch((size_t)BT * Hh + 64);   // ln1 out; reused for ln2 out
    _Float16* q16     = alloch((size_t)BT * Hh + 64);
    _Float16* k16     = alloch((size_t)BT * Hh + 64);
    _Float16* v16     = alloch((size_t)BT * Hh + 64);
    _Float16* att16   = alloch((size_t)Bb * HEADS * T * T + 64);
    _Float16* o16     = alloch((size_t)BT * Hh + 64);
    _Float16* ffh16   = alloch((size_t)BT * Ff + 64);
    _Float16* Wq16    = alloch((size_t)L * Hh * Hh);
    _Float16* Wk16    = alloch((size_t)L * Hh * Hh);
    _Float16* Wv16    = alloch((size_t)L * Hh * Hh);
    _Float16* Wo16    = alloch((size_t)L * Hh * Hh);
    _Float16* W116    = alloch((size_t)L * Hh * Ff);
    _Float16* W216    = alloch((size_t)L * Ff * Hh);

    // ---- one-time conversions ----
    agltsa_zero16<<<1, 64, 0, stream>>>(zb16, 64);
    agltsa_gram<<<(Nn * NP + 255) / 256, 256, 0, stream>>>(emb, gram16, Nn, E, NP);
    agltsa_cvt_pad<<<((L * Nn) * NP + 255) / 256, 256, 0, stream>>>(aglW, aglW16, L * Nn, Nn, NP);
    {
        long long nw = (long long)L * Hh * Hh;
        agltsa_cvt<<<(int)((nw + 255) / 256), 256, 0, stream>>>(Wq, Wq16, nw);
        agltsa_cvt<<<(int)((nw + 255) / 256), 256, 0, stream>>>(Wk, Wk16, nw);
        agltsa_cvt<<<(int)((nw + 255) / 256), 256, 0, stream>>>(Wv, Wv16, nw);
        agltsa_cvt<<<(int)((nw + 255) / 256), 256, 0, stream>>>(Wo, Wo16, nw);
        long long nf = (long long)L * Hh * Ff;
        agltsa_cvt<<<(int)((nf + 255) / 256), 256, 0, stream>>>(W1, W116, nf);
        agltsa_cvt<<<(int)((nf + 255) / 256), 256, 0, stream>>>(W2, W216, nf);
    }

    for (int l = 0; l < L; ++l) {
        const float* xsrc = (l == 0) ? inputs : xbuf;

        // A = softmax(relu(gram @ W_l + b_l))
        launch_gemm2d(stream, gram16, aglW16 + (size_t)l * Nn * NP, Aadj, nullptr,
                      aglB + (size_t)l * Nn, nullptr,
                      Nn, Nn, Nn, NP, NP, NP, NP, Nn, 0, 1.0f, 1, zb16);
        agltsa_softmax_rows<<<Nn, 256, 0, stream>>>(Aadj, Aadj16, Nn, Nn, NP);

        // T2 = 2*(A@A) - I
        launch_gemm2d(stream, Aadj16, Aadj16, T2f, nullptr, nullptr, nullptr,
                      Nn, Nn, Nn, NP, NP, NP, NP, Nn, 0, 2.0f, 0, zb16);
        agltsa_cheb_fix<<<(Nn * NP + 255) / 256, 256, 0, stream>>>(T2f, T216, Nn, NP);

        // x -> Xt16 (N x BT*C)
        {
            long long tot = (long long)BT * Nn * Cc;
            agltsa_transpose_x<<<(int)((tot + 255) / 256), 256, 0, stream>>>(xsrc, Xt16, BT, Nn, Cc);
        }
        // graph conv: Yk = S_k @ Xt  (S0 = I handled in glout)
        launch_gemm2d(stream, Aadj16, Xt16, Y1, nullptr, nullptr, nullptr,
                      Nn, KC, Nn, NP, KC, NP, KC, KC, 0, 1.0f, 0, zb16);
        launch_gemm2d(stream, T216, Xt16, Y2, nullptr, nullptr, nullptr,
                      Nn, KC, Nn, NP, KC, NP, KC, KC, 0, 1.0f, 0, zb16);
        {
            long long tot = (long long)BT * Nn;
            agltsa_glout<<<(int)((tot + 255) / 256), 256, 0, stream>>>(
                xsrc, Y1, Y2, aoW + (size_t)l * 48, aoB + (size_t)l * 4,
                gbuf, BT, Nn, T, (l == 0) ? 1 : 0);
        }

        // y = LN1(g)  (f16 only, feeds QKV)
        agltsa_layernorm<<<BT, 256, 0, stream>>>(gbuf, ln1g + (size_t)l * Hh,
                                                 ln1b + (size_t)l * Hh, nullptr, y16, Hh);
        // q (pre-scaled) / k / v  (f16 outputs)
        launch_gemm2d(stream, y16, Wq16 + (size_t)l * Hh * Hh, nullptr, q16, nullptr, nullptr,
                      BT, Hh, Hh, Hh, Hh, Hh, Hh, Hh, 0, scale, 0, zb16);
        launch_gemm2d(stream, y16, Wk16 + (size_t)l * Hh * Hh, nullptr, k16, nullptr, nullptr,
                      BT, Hh, Hh, Hh, Hh, Hh, Hh, Hh, 0, 1.0f, 0, zb16);
        launch_gemm2d(stream, y16, Wv16 + (size_t)l * Hh * Hh, nullptr, v16, nullptr, nullptr,
                      BT, Hh, Hh, Hh, Hh, Hh, Hh, Hh, 0, 1.0f, 0, zb16);

        // scores = q @ k^T per (b,h), then softmax (f32 in place + f16 copy)
        launch_gemm(stream, q16, k16, attf, nullptr, nullptr, nullptr,
                    T, T, DK, DK, 0, Hh, Hh, T, 0,
                    (long long)T * Hh, DK, (long long)T * Hh, DK,
                    (long long)HEADS * T * T, (long long)T * T, 0, 0,
                    Bb * HEADS, HEADS, 1.0f, 0, /*transB=*/1, zb16);
        agltsa_softmax_rows<<<Bb * HEADS * T, 128, 0, stream>>>(attf, att16, Bb * HEADS * T, T, T);

        // o = att @ v  (f16 out, heads-major repack via strides)
        launch_gemm(stream, att16, v16, nullptr, o16, nullptr, nullptr,
                    T, DK, T, T, 1 << 28, T, Hh, Hh, 0,
                    (long long)HEADS * T * T, (long long)T * T,
                    (long long)T * Hh, DK,
                    (long long)T * Hh, DK, 0, 0,
                    Bb * HEADS, HEADS, 1.0f, 0, 0, zb16);

        // g2 = g + o @ Wo
        launch_gemm2d(stream, o16, Wo16 + (size_t)l * Hh * Hh, g2, nullptr, nullptr, gbuf,
                      BT, Hh, Hh, Hh, Hh, Hh, Hh, Hh, Hh, 1.0f, 0, zb16);

        // FFN with pre-norm + residual
        agltsa_layernorm<<<BT, 256, 0, stream>>>(g2, ln2g + (size_t)l * Hh,
                                                 ln2b + (size_t)l * Hh, nullptr, y16, Hh);
        launch_gemm2d(stream, y16, W116 + (size_t)l * Hh * Ff, nullptr, ffh16,
                      b1 + (size_t)l * Ff, nullptr,
                      BT, Ff, Hh, Hh, Ff, Hh, Ff, Ff, 0, 1.0f, 1, zb16);
        launch_gemm2d(stream, ffh16, W216 + (size_t)l * Ff * Hh, xbuf, nullptr,
                      b2 + (size_t)l * Hh, g2,
                      BT, Hh, Ff, Ff, Hh, Ff, Hh, Hh, Hh, 1.0f, 0, zb16);
    }

    // final layernorm -> d_out (f32)
    agltsa_layernorm<<<BT, 256, 0, stream>>>(xbuf, lastg, lastb, (float*)d_out, nullptr, Hh);
}
